// MambaEncoder_59785944761168
// MI455X (gfx1250) — compile-verified, matrix operand order checked by reading, not run on hardware
//
#include <hip/hip_runtime.h>
#include <math.h>

// ---------------------------------------------------------------------------
// MI455X (gfx1250, CDNA5, wave32) implementation of the SegMamba-style
// encoder.  All dense contractions go through a bf16 WMMA GEMM
// (v_wmma_f32_16x16x32_bf16) with fused epilogues; the selective scan is a
// register-resident 16-state recurrence with async global->LDS staging of
// the shared B/C state matrices; 3D convs are direct.
// ---------------------------------------------------------------------------

#define LRELU_K 0.01f
#define EPS_K   1e-5f
#define BN_RSQ  0.9999950000375f      // 1/sqrt(1+1e-5)

typedef __attribute__((ext_vector_type(16))) __bf16 v16bf;
typedef __attribute__((ext_vector_type(8)))  float  v8f;

// ---- async global->LDS copy (CDNA5 GLOBAL_LOAD_ASYNC_TO_LDS_B32) ----
#if __has_builtin(__builtin_amdgcn_global_load_async_to_lds_b32)
#define ASYNC_G2L_B32(g, lds)                                              \
  __builtin_amdgcn_global_load_async_to_lds_b32(                           \
      (__attribute__((address_space(1))) int*)(g),                         \
      (__attribute__((address_space(3))) int*)(lds), 0, 0)
#if __has_builtin(__builtin_amdgcn_s_wait_asynccnt)
#define ASYNC_WAIT() __builtin_amdgcn_s_wait_asynccnt(0)
#else
#define ASYNC_WAIT() asm volatile("s_wait_asynccnt 0" ::: "memory")
#endif
#else
#define ASYNC_G2L_B32(g, lds) (*(lds) = *(g))
#define ASYNC_WAIT()
#endif

static __device__ __forceinline__ float apply_act(float v, int act) {
  switch (act) {
    case 1: return v >= 0.f ? v : LRELU_K * v;                   // lrelu
    case 2: return v / (1.f + __expf(-v));                       // silu
    case 3: return 0.5f * v * (1.f + erff(v * 0.70710678f));     // gelu(exact)
    case 4: return v > 20.f ? v : log1pf(__expf(v));             // softplus
    default: return v;
  }
}

// ---------------------------------------------------------------------------
// Tiled WMMA GEMM:  C[m,n] = act( sum_k A[m,k]*B[k,n] + bias1[m] ) * scale[m]
//                            * sconst + bias2[m]
// A row-major [M,K] (lda), B generic strides (sbk, sbn), C row-major (ldc).
// Block = 128 threads = 4 waves; block tile 64x64; wave tile 32x32
// (2x2 fragments of 16x16); K-step 32.
//
// LDS tiles use a fragment-ready permuted layout (CDNA5 ISA 7.12.2, 16-bit
// operands): lane (row, hi) reads its 16 bf16 halves as one contiguous,
// 32B-aligned region -> plain v16bf load (2x ds_load_b128).  The loader
// packs two f32->bf16 per dword and issues single ds_store_b32's.
//
// Boundary handling is BLOCK-UNIFORM: interior blocks (full 64x64 M/N tile)
// take a fast path whose global loads carry no per-lane predication at all
// (scalar s_cbranch only), so they clause under one wait.  Edge blocks and
// the K-tail step use a clamped+masked variant.
// ---------------------------------------------------------------------------
__global__ __launch_bounds__(128) void gemm_wmma_k(
    const float* __restrict__ A, const float* __restrict__ B,
    float* __restrict__ Cout, int M, int N, int K, int lda,
    long long sbk, long long sbn, int ldc,
    const float* __restrict__ bias1, const float* __restrict__ scale,
    const float* __restrict__ bias2, float sconst, int act) {
  // row stride 48 halves = 96 bytes (each hi-half region is 32B aligned)
  __shared__ __attribute__((aligned(32))) __bf16 As2[64 * 48];
  __shared__ __attribute__((aligned(32))) __bf16 Bs2[64 * 48];
  unsigned* Asd = (unsigned*)As2;   // dword view, row stride 24
  unsigned* Bsd = (unsigned*)Bs2;

  const int tid  = threadIdx.x;
  const int lane = tid & 31;
  const int wid  = tid >> 5;
  const int m0 = blockIdx.y * 64, n0 = blockIdx.x * 64;
  const int wm = (wid >> 1) * 32, wn = (wid & 1) * 32;
  const int fr = lane & 15, hi = lane >> 4;

  // per-thread staging coordinates (fixed across K steps)
  const int arow = tid >> 4, ak0 = (tid & 15) * 2;           // A: 64r x 16 kp
  const int bn_  = tid & 63, bk0 = (tid >> 6) * 2;           // B: lanes on n
  const int ahh = (ak0 >> 3) & 1, app = (ak0 & 7) + ((ak0 >> 4) << 3);

  v8f acc00 = {}, acc01 = {}, acc10 = {}, acc11 = {};

  const bool full = (m0 + 64 <= M) && (n0 + 64 <= N);
  const int Kmain = K & ~31;

// ---- unconditional staging: requires kk+32 <= K and full M/N tile ----
#define STAGE_FULL(KK)                                                        \
  {                                                                           \
    _Pragma("unroll")                                                         \
    for (int i = 0; i < 8; ++i) {                                             \
      int gm = m0 + arow + i * 8;                                             \
      float a0 = A[(long long)gm * lda + (KK) + ak0];                         \
      float a1 = A[(long long)gm * lda + (KK) + ak0 + 1];                     \
      union { unsigned u; __bf16 h[2]; } pk;                                  \
      pk.h[0] = (__bf16)a0; pk.h[1] = (__bf16)a1;                             \
      Asd[(arow + i * 8) * 24 + ahh * 8 + (app >> 1)] = pk.u;                 \
    }                                                                         \
    _Pragma("unroll")                                                         \
    for (int i = 0; i < 8; ++i) {                                             \
      long long cn = (long long)(n0 + bn_) * sbn;                             \
      int gk = (KK) + bk0 + i * 4;                                            \
      float b0 = B[(long long)gk * sbk + cn];                                 \
      float b1 = B[(long long)(gk + 1) * sbk + cn];                           \
      union { unsigned u; __bf16 h[2]; } pk;                                  \
      pk.h[0] = (__bf16)b0; pk.h[1] = (__bf16)b1;                             \
      int kq = bk0 + i * 4;                                                   \
      int hh = (kq >> 3) & 1, pp = (kq & 7) + ((kq >> 4) << 3);               \
      Bsd[bn_ * 24 + hh * 8 + (pp >> 1)] = pk.u;                              \
    }                                                                         \
  }

// ---- masked staging: clamped addresses + cndmask (edge blocks / K tail) ---
#define STAGE_MASKED(KK)                                                      \
  {                                                                           \
    _Pragma("unroll")                                                         \
    for (int i = 0; i < 8; ++i) {                                             \
      int row = arow + i * 8;                                                 \
      int gm = m0 + row, gk = (KK) + ak0;                                     \
      int cm = gm < M ? gm : M - 1;                                           \
      int c0 = gk < K ? gk : K - 1;                                           \
      int c1 = gk + 1 < K ? gk + 1 : K - 1;                                   \
      float a0 = A[(long long)cm * lda + c0];                                 \
      float a1 = A[(long long)cm * lda + c1];                                 \
      a0 = (gm < M && gk < K) ? a0 : 0.f;                                     \
      a1 = (gm < M && gk + 1 < K) ? a1 : 0.f;                                 \
      union { unsigned u; __bf16 h[2]; } pk;                                  \
      pk.h[0] = (__bf16)a0; pk.h[1] = (__bf16)a1;                             \
      Asd[row * 24 + ahh * 8 + (app >> 1)] = pk.u;                            \
    }                                                                         \
    _Pragma("unroll")                                                         \
    for (int i = 0; i < 8; ++i) {                                             \
      int kq = bk0 + i * 4;                                                   \
      int gn = n0 + bn_, gk = (KK) + kq;                                      \
      long long cn = (long long)(gn < N ? gn : N - 1) * sbn;                  \
      int c0 = gk < K ? gk : K - 1;                                           \
      int c1 = gk + 1 < K ? gk + 1 : K - 1;                                   \
      float b0 = B[(long long)c0 * sbk + cn];                                 \
      float b1 = B[(long long)c1 * sbk + cn];                                 \
      b0 = (gn < N && gk < K) ? b0 : 0.f;                                     \
      b1 = (gn < N && gk + 1 < K) ? b1 : 0.f;                                 \
      union { unsigned u; __bf16 h[2]; } pk;                                  \
      pk.h[0] = (__bf16)b0; pk.h[1] = (__bf16)b1;                             \
      int hh = (kq >> 3) & 1, pp = (kq & 7) + ((kq >> 4) << 3);               \
      Bsd[bn_ * 24 + hh * 8 + (pp >> 1)] = pk.u;                              \
    }                                                                         \
  }

// ---- one 32-deep WMMA step from the fragment-ready LDS tiles ----
#define COMPUTE_STEP()                                                        \
  {                                                                           \
    v16bf af0 = *(const v16bf*)&As2[(wm + fr) * 48 + hi * 16];                \
    v16bf af1 = *(const v16bf*)&As2[(wm + 16 + fr) * 48 + hi * 16];           \
    v16bf bf0 = *(const v16bf*)&Bs2[(wn + fr) * 48 + hi * 16];                \
    v16bf bf1 = *(const v16bf*)&Bs2[(wn + 16 + fr) * 48 + hi * 16];           \
    acc00 = __builtin_amdgcn_wmma_f32_16x16x32_bf16(false, af0, false, bf0,   \
                                                    (short)0, acc00, false, false); \
    acc01 = __builtin_amdgcn_wmma_f32_16x16x32_bf16(false, af0, false, bf1,   \
                                                    (short)0, acc01, false, false); \
    acc10 = __builtin_amdgcn_wmma_f32_16x16x32_bf16(false, af1, false, bf0,   \
                                                    (short)0, acc10, false, false); \
    acc11 = __builtin_amdgcn_wmma_f32_16x16x32_bf16(false, af1, false, bf1,   \
                                                    (short)0, acc11, false, false); \
  }

  if (full) {
    for (int kk = 0; kk < Kmain; kk += 32) {
      STAGE_FULL(kk);
      if (kk + 32 < K) {
        int gm = m0 + (tid >> 1);
        __builtin_prefetch(&A[(long long)gm * lda + kk + 32], 0, 1);
      }
      __syncthreads();
      COMPUTE_STEP();
      __syncthreads();
    }
  } else {
    for (int kk = 0; kk < Kmain; kk += 32) {
      STAGE_MASKED(kk);
      __syncthreads();
      COMPUTE_STEP();
      __syncthreads();
    }
  }
  if (Kmain < K) {               // K tail (K not a multiple of 32)
    STAGE_MASKED(Kmain);
    __syncthreads();
    COMPUTE_STEP();
    __syncthreads();
  }

  // ---- fused epilogue ----
  float* accp[4] = {(float*)&acc00, (float*)&acc01, (float*)&acc10,
                    (float*)&acc11};
#pragma unroll
  for (int i = 0; i < 2; ++i)
#pragma unroll
    for (int j = 0; j < 2; ++j) {
      float* av = accp[i * 2 + j];
#pragma unroll
      for (int r = 0; r < 8; ++r) {
        int m = m0 + wm + i * 16 + r + 8 * hi;  // C layout: VGPR r -> M=r/r+8
        int n = n0 + wn + j * 16 + (lane & 15);
        if (full || (m < M && n < N)) {
          float v = av[r];
          if (bias1) v += bias1[m];
          v = apply_act(v, act);
          if (scale) v = v * (scale[m] * sconst) + (bias2 ? bias2[m] : 0.f);
          Cout[(long long)m * ldc + n] = v;
        }
      }
    }
#undef STAGE_FULL
#undef STAGE_MASKED
#undef COMPUTE_STEP
}

// ---------------------------------------------------------------------------
// Direct 3D conv (cubic kernel KD, stride, pad) with optional fused BN + lrelu
// ---------------------------------------------------------------------------
__global__ void conv3d_direct_k(const float* __restrict__ in,
                                const float* __restrict__ w,
                                const float* __restrict__ bias,
                                const float* __restrict__ bnG,
                                const float* __restrict__ bnB, int lrelu,
                                float* __restrict__ out, int Cin, int Cout,
                                int D, int H, int W, int KD, int stride,
                                int pad, int Din, int Hin, int Win) {
  int idx = blockIdx.x * blockDim.x + threadIdx.x;
  int total = Cout * D * H * W;
  if (idx >= total) return;
  int x = idx % W, t = idx / W;
  int y = t % H;  t /= H;
  int z = t % D;
  int co = t / D;
  float acc = bias[co];
  for (int ci = 0; ci < Cin; ++ci)
    for (int kz = 0; kz < KD; ++kz) {
      int zi = z * stride - pad + kz;
      if (zi < 0 || zi >= Din) continue;
      for (int ky = 0; ky < KD; ++ky) {
        int yi = y * stride - pad + ky;
        if (yi < 0 || yi >= Hin) continue;
        for (int kx = 0; kx < KD; ++kx) {
          int xi = x * stride - pad + kx;
          if (xi < 0 || xi >= Win) continue;
          acc += in[((size_t)(ci * Din + zi) * Hin + yi) * Win + xi] *
                 w[(((size_t)(co * Cin + ci) * KD + kz) * KD + ky) * KD + kx];
        }
      }
    }
  if (bnG) acc = acc * (bnG[co] * BN_RSQ) + bnB[co];
  if (lrelu) acc = acc >= 0.f ? acc : LRELU_K * acc;
  out[idx] = acc;
}

// out = (a + b) * g[c]*BN_RSQ + bb[c]   (g may be null -> plain add)
__global__ void add_scale_bias_k(float* __restrict__ out,
                                 const float* __restrict__ a,
                                 const float* __restrict__ b,
                                 const float* __restrict__ g,
                                 const float* __restrict__ bb, int C, int L) {
  int idx = blockIdx.x * blockDim.x + threadIdx.x;
  if (idx >= C * L) return;
  int c = idx / L;
  float v = a[idx] + b[idx];
  if (g) v = v * (g[c] * BN_RSQ) + bb[c];
  out[idx] = v;
}

// per-channel instance norm over L (one workgroup per channel)
__global__ void instancenorm_k(const float* __restrict__ x,
                               float* __restrict__ out, int C, int L) {
  int c = blockIdx.x;
  const float* xc = x + (size_t)c * L;
  __shared__ float s1[256], s2[256];
  float a = 0.f, b = 0.f;
  for (int l = threadIdx.x; l < L; l += blockDim.x) {
    float v = xc[l];
    a += v; b += v * v;
  }
  s1[threadIdx.x] = a; s2[threadIdx.x] = b;
  __syncthreads();
  for (int st = 128; st > 0; st >>= 1) {
    if ((int)threadIdx.x < st) {
      s1[threadIdx.x] += s1[threadIdx.x + st];
      s2[threadIdx.x] += s2[threadIdx.x + st];
    }
    __syncthreads();
  }
  float mu = s1[0] / (float)L;
  float var = s2[0] / (float)L - mu * mu;
  float inv = rsqrtf(var + EPS_K);
  for (int l = threadIdx.x; l < L; l += blockDim.x)
    out[(size_t)c * L + l] = (xc[l] - mu) * inv;
}

// layernorm across channels (x is [C,L] channel-major; one thread per l)
__global__ void layernorm_cl_k(const float* __restrict__ x,
                               const float* __restrict__ g,
                               const float* __restrict__ b,
                               float* __restrict__ out, int C, int L) {
  int l = blockIdx.x * blockDim.x + threadIdx.x;
  if (l >= L) return;
  float mu = 0.f;
  for (int c = 0; c < C; ++c) mu += x[(size_t)c * L + l];
  mu /= (float)C;
  float var = 0.f;
  for (int c = 0; c < C; ++c) {
    float d = x[(size_t)c * L + l] - mu;
    var += d * d;
  }
  var /= (float)C;
  float inv = rsqrtf(var + EPS_K);
  for (int c = 0; c < C; ++c)
    out[(size_t)c * L + l] = (x[(size_t)c * L + l] - mu) * inv * g[c] + b[c];
}

// causal depthwise conv1d (K=4) + SiLU
__global__ void causal_conv1d_silu_k(const float* __restrict__ x,
                                     const float* __restrict__ w,
                                     const float* __restrict__ b,
                                     float* __restrict__ out, int Dn, int L) {
  int idx = blockIdx.x * blockDim.x + threadIdx.x;
  if (idx >= Dn * L) return;
  int d = idx / L, l = idx % L;
  float acc = b[d];
#pragma unroll
  for (int k = 0; k < 4; ++k) {
    int j = l - 3 + k;
    if (j >= 0) acc += w[d * 4 + k] * x[(size_t)d * L + j];
  }
  out[idx] = acc / (1.f + __expf(-acc));
}

// row gather / scatter-add for sequence reversal and slice interleave.
// mode: 0 identity, 1 reverse, 2 slice-fwd (src=(l%ns)*m + l/ns),
//       3 slice-inv (src=(l%m)*ns + l/m).  accum: out += instead of =.
__global__ void permute_rows_k(const float* __restrict__ in,
                               float* __restrict__ out, int E, int L, int ns,
                               int m, int mode, int accum) {
  int idx = blockIdx.x * blockDim.x + threadIdx.x;
  if (idx >= E * L) return;
  int d = idx / L, l = idx % L;
  int src;
  switch (mode) {
    case 1:  src = L - 1 - l; break;
    case 2:  src = (l % ns) * m + l / ns; break;
    case 3:  src = (l % m) * ns + l / m; break;
    default: src = l; break;
  }
  float v = in[(size_t)d * L + src];
  if (accum) out[idx] += v; else out[idx] = v;
}

// ---------------------------------------------------------------------------
// Selective scan: 16-state recurrence per channel (registers).  The B_/C_
// matrices [16][L] are shared by every channel, so they are staged chunk-wise
// into LDS with CDNA5 async global->LDS copies (GLOBAL_LOAD_ASYNC_TO_LDS_B32,
// tracked by ASYNCcnt) and then read as conflict-free LDS broadcasts.
// All threads participate in staging/barriers; inactive channels only skip
// the final store.
// ---------------------------------------------------------------------------
#define SCAN_CH 64
__global__ __launch_bounds__(128) void selective_scan_k(
    const float* __restrict__ u, const float* __restrict__ dl,
    const float* __restrict__ A_log, const float* __restrict__ Bm,
    const float* __restrict__ Cm, const float* __restrict__ Dp,
    const float* __restrict__ z, float* __restrict__ out, int di, int L) {
  __shared__ float BmS[16][SCAN_CH];
  __shared__ float CmS[16][SCAN_CH];

  int d = blockIdx.x * blockDim.x + threadIdx.x;
  const bool active = d < di;
  const int dc = active ? d : di - 1;     // clamped channel for safe loads

  float Aa[16], h[16];
#pragma unroll
  for (int n = 0; n < 16; ++n) {
    Aa[n] = -__expf(A_log[dc * 16 + n]);
    h[n] = 0.f;
  }
  float Dv = Dp[dc];
  const float* ud = u  + (size_t)dc * L;
  const float* dd = dl + (size_t)dc * L;
  const float* zd = z  + (size_t)dc * L;
  float* od = out + (size_t)dc * L;

  const int srow = threadIdx.x >> 6;      // staging: 2 rows per pass
  const int scol = threadIdx.x & 63;

  for (int l0 = 0; l0 < L; l0 += SCAN_CH) {
    __syncthreads();                      // previous chunk fully consumed
    int gc = l0 + scol < L ? l0 + scol : L - 1;
#pragma unroll
    for (int i = 0; i < 8; ++i) {         // 16 rows x 64 cols, 2 rows/pass
      int nrow = srow + i * 2;
      ASYNC_G2L_B32(&Bm[(size_t)nrow * L + gc], &BmS[nrow][scol]);
      ASYNC_G2L_B32(&Cm[(size_t)nrow * L + gc], &CmS[nrow][scol]);
    }
    ASYNC_WAIT();
    __syncthreads();

    int lend = (L - l0 < SCAN_CH) ? (L - l0) : SCAN_CH;
    for (int ll = 0; ll < lend; ++ll) {
      int l = l0 + ll;
      float de = dd[l], uv = ud[l];
      float y = Dv * uv;
      float dbu = de * uv;
#pragma unroll
      for (int n = 0; n < 16; ++n) {
        h[n] = h[n] * __expf(de * Aa[n]) + dbu * BmS[n][ll];
        y += h[n] * CmS[n][ll];
      }
      float zz = zd[l];
      if (active) od[l] = y * (zz / (1.f + __expf(-zz)));
    }
  }
}

// ---------------------------------------------------------------------------
// Host orchestration
// ---------------------------------------------------------------------------
struct MDirP {
  const float *A_log, *D, *conv_b, *conv_w, *dt_proj_b, *dt_proj_w, *x_proj_w;
};
struct LayerP {
  const float *bn2_b, *bn2_g, *ln_b, *ln_g;
  MDirP bwd, fwd, sli;
  const float *in_proj_w, *out_proj_w;
};
struct GscP {
  const float *b1, *b2, *b3, *b4, *bn1_b, *bn1_g, *bn3_b, *bn3_g, *w1, *w2,
      *w3, *w4;
};
struct MlpP {
  const float *b1, *b2, *bn_b, *bn_g, *w1, *w2;
};

static inline int cdiv(int a, int b) { return (a + b - 1) / b; }

extern "C" void kernel_launch(void* const* d_in, const int* in_sizes, int n_in,
                              void* d_out, int out_size, void* d_ws,
                              size_t ws_size, hipStream_t stream) {
  (void)in_sizes; (void)n_in; (void)out_size; (void)ws_size;

  // ---- parameter unflatten (jax tree order: sorted dict keys) ----
  // top: 'down','gsc','mlp','stage'; x is d_in[0].
  int t = 1;
  auto F = [&](void) { return (const float*)d_in[t++]; };
  const float *down_b[4], *down_w[4];
  for (int i = 0; i < 4; ++i) { down_b[i] = F(); down_w[i] = F(); }
  GscP gsc[4];
  for (int i = 0; i < 4; ++i) {
    gsc[i].b1 = F(); gsc[i].b2 = F(); gsc[i].b3 = F(); gsc[i].b4 = F();
    gsc[i].bn1_b = F(); gsc[i].bn1_g = F();
    gsc[i].bn3_b = F(); gsc[i].bn3_g = F();
    gsc[i].w1 = F(); gsc[i].w2 = F(); gsc[i].w3 = F(); gsc[i].w4 = F();
  }
  MlpP mlp[4];
  for (int i = 0; i < 4; ++i) {
    mlp[i].b1 = F(); mlp[i].b2 = F();
    mlp[i].bn_b = F(); mlp[i].bn_g = F();
    mlp[i].w1 = F(); mlp[i].w2 = F();
  }
  auto readMDir = [&](MDirP& m) {
    m.A_log = F(); m.D = F(); m.conv_b = F(); m.conv_w = F();
    m.dt_proj_b = F(); m.dt_proj_w = F(); m.x_proj_w = F();
  };
  LayerP layers[4][2];
  for (int i = 0; i < 4; ++i)
    for (int j = 0; j < 2; ++j) {
      LayerP& lp = layers[i][j];
      lp.bn2_b = F(); lp.bn2_g = F(); lp.ln_b = F(); lp.ln_g = F();
      readMDir(lp.bwd); readMDir(lp.fwd);
      lp.in_proj_w = F(); lp.out_proj_w = F();
      readMDir(lp.sli);
    }

  // ---- workspace layout (floats) ----
  float* ws = (float*)d_ws;
  float* X     = ws;
  float* T1    = X    + 786432;
  float* T2    = T1   + 786432;
  float* LNb   = T2   + 786432;
  float* XZ    = LNb  + 786432;
  float* XZP   = XZ   + 3145728;
  float* U     = XZP  + 3145728;
  float* XDBL  = U    + 1572864;
  float* DELTA = XDBL + 573440;
  float* Y     = DELTA + 1572864;
  float* ACC   = Y    + 1572864;
  float* MLPH  = ACC  + 1572864;

  const int DIMS[4] = {48, 96, 192, 384};
  const int NSL[4]  = {64, 32, 16, 8};
  const int SD[4] = {32, 16, 8, 4}, SH[4] = {32, 16, 8, 4}, SW[4] = {16, 8, 4, 2};

  auto gemm = [&](const float* A, const float* B, float* C, int M, int N,
                  int K, int lda, long long sbk, long long sbn, int ldc,
                  const float* b1, const float* sc, const float* b2,
                  float sconst, int act) {
    dim3 g(cdiv(N, 64), cdiv(M, 64));
    gemm_wmma_k<<<g, dim3(128), 0, stream>>>(A, B, C, M, N, K, lda, sbk, sbn,
                                             ldc, b1, sc, b2, sconst, act);
  };

  auto run_branch = [&](const float* xz, const MDirP& md, float* outp, int di,
                        int r, int L) {
    int n = di * L;
    causal_conv1d_silu_k<<<cdiv(n, 256), 256, 0, stream>>>(
        xz, md.conv_w, md.conv_b, U, di, L);
    // x_dbl = x_proj_w [(r+32),di] x U [di,L]
    gemm(md.x_proj_w, U, XDBL, r + 32, L, di, di, (long long)L, 1LL, L,
         nullptr, nullptr, nullptr, 1.f, 0);
    // delta = softplus(dt_proj_w [di,r] x x_dbl[0:r] + dt_proj_b)
    gemm(md.dt_proj_w, XDBL, DELTA, di, L, r, r, (long long)L, 1LL, L,
         md.dt_proj_b, nullptr, nullptr, 1.f, 4);
    selective_scan_k<<<cdiv(di, 128), 128, 0, stream>>>(
        U, DELTA, md.A_log, XDBL + (size_t)r * L, XDBL + (size_t)(r + 16) * L,
        md.D, xz + (size_t)di * L, outp, di, L);
  };

  const float* input = (const float*)d_in[0];   // (1,1,64,64,32) f32
  float* outp = (float*)d_out;
  size_t out_off = 0;
  int Dp = 64, Hp = 64, Wp = 32;

  for (int s = 0; s < 4; ++s) {
    const int C = DIMS[s], D = SD[s], H = SH[s], W = SW[s];
    const int L = D * H * W, di = 2 * C, r = (C + 15) / 16;
    const int ns = NSL[s], mlen = L / ns;
    const int CL = C * L;

    // ---- downsample ----
    if (s == 0) {
      conv3d_direct_k<<<cdiv(CL, 256), 256, 0, stream>>>(
          input, down_w[0], down_b[0], nullptr, nullptr, 0, X, 1, C, D, H, W,
          7, 2, 3, 64, 64, 32);
    } else {
      int Cp = DIMS[s - 1], Lp = Dp * Hp * Wp;
      instancenorm_k<<<Cp, 256, 0, stream>>>(X, T1, Cp, Lp);
      conv3d_direct_k<<<cdiv(CL, 256), 256, 0, stream>>>(
          T1, down_w[s], down_b[s], nullptr, nullptr, 0, X, Cp, C, D, H, W, 2,
          2, 0, Dp, Hp, Wp);
    }

    // ---- GSC block ----
    const GscP& g = gsc[s];
    conv3d_direct_k<<<cdiv(CL, 256), 256, 0, stream>>>(
        X, g.w1, g.b1, g.bn1_g, g.bn1_b, 1, T1, C, C, D, H, W, 3, 1, 1, D, H, W);
    conv3d_direct_k<<<cdiv(CL, 256), 256, 0, stream>>>(
        T1, g.w2, g.b2, nullptr, nullptr, 1, T2, C, C, D, H, W, 3, 1, 1, D, H, W);
    gemm(g.w3, X, LNb, C, L, C, C, (long long)L, 1LL, L, g.b3, nullptr,
         nullptr, 1.f, 0);                               // x2 = 1x1 conv
    add_scale_bias_k<<<cdiv(CL, 256), 256, 0, stream>>>(T1, T2, LNb, g.bn3_g,
                                                        g.bn3_b, C, L);
    gemm(g.w4, T1, T2, C, L, C, C, (long long)L, 1LL, L, g.b4, nullptr,
         nullptr, 1.f, 0);
    add_scale_bias_k<<<cdiv(CL, 256), 256, 0, stream>>>(X, T2, X, nullptr,
                                                        nullptr, C, L);

    // ---- Mamba layers ----
    for (int j = 0; j < 2; ++j) {
      const LayerP& lp = layers[s][j];
      layernorm_cl_k<<<cdiv(L, 256), 256, 0, stream>>>(X, lp.ln_g, lp.ln_b,
                                                       LNb, C, L);
      gemm(lp.in_proj_w, LNb, XZ, 2 * di, L, C, C, (long long)L, 1LL, L,
           nullptr, nullptr, nullptr, 1.f, 0);
      // forward branch -> ACC
      run_branch(XZ, lp.fwd, ACC, di, r, L);
      // backward branch: reverse, run, reversed add
      permute_rows_k<<<cdiv(2 * di * L, 256), 256, 0, stream>>>(
          XZ, XZP, 2 * di, L, ns, mlen, 1, 0);
      run_branch(XZP, lp.bwd, Y, di, r, L);
      permute_rows_k<<<cdiv(di * L, 256), 256, 0, stream>>>(
          Y, ACC, di, L, ns, mlen, 1, 1);
      // slice branch: interleave, run, inverse-interleave add
      permute_rows_k<<<cdiv(2 * di * L, 256), 256, 0, stream>>>(
          XZ, XZP, 2 * di, L, ns, mlen, 2, 0);
      run_branch(XZP, lp.sli, Y, di, r, L);
      permute_rows_k<<<cdiv(di * L, 256), 256, 0, stream>>>(
          Y, ACC, di, L, ns, mlen, 3, 1);
      // out_proj + residual + bn2
      gemm(lp.out_proj_w, ACC, T1, C, L, di, di, (long long)L, 1LL, L,
           nullptr, nullptr, nullptr, 1.f, 0);
      add_scale_bias_k<<<cdiv(CL, 256), 256, 0, stream>>>(X, X, T1, lp.bn2_g,
                                                          lp.bn2_b, C, L);
    }

    // ---- MLP head -> d_out ----
    const MlpP& mp = mlp[s];
    instancenorm_k<<<C, 256, 0, stream>>>(X, T1, C, L);
    gemm(mp.w1, T1, MLPH, 2 * C, L, C, C, (long long)L, 1LL, L, mp.b1,
         nullptr, nullptr, 1.f, 3);                           // gelu
    gemm(mp.w2, MLPH, outp + out_off, C, L, 2 * C, 2 * C, (long long)L, 1LL,
         L, mp.b2, mp.bn_g, mp.bn_b, BN_RSQ, 0);              // + bn
    out_off += (size_t)CL;

    Dp = D; Hp = H; Wp = W;
  }
}
